// CausalSelfAttention_21672404976289
// MI455X (gfx1250) — matrix operand order repaired and checked by reading, not compiled, not run on hardware
//
#include <hip/hip_runtime.h>

#define Bn   4
#define Tn   2048
#define Cn   1024
#define NHn  16
#define HDn  64
#define C3n  (3 * Cn)
#define BT   (Bn * Tn)

// GEMM blocking
#define BM 64
#define BN 64
#define BK 32

typedef _Float16 v16h __attribute__((ext_vector_type(16)));
typedef float    v8f  __attribute__((ext_vector_type(8)));

union Frag16 { v16h v; _Float16 h[16]; };
union FragC  { v8f  v; float     f[8]; };

__device__ __forceinline__ v8f wmma16(v16h a, v16h b, v8f c) {
  // D = A(16x32 f16) * B(32x16 f16) + C(16x16 f32)
  return __builtin_amdgcn_wmma_f32_16x16x32_f16(
      /*neg_a=*/false, a, /*neg_b=*/false, b,
      /*c_mod=*/(short)0, c, /*reuse_a=*/false, /*reuse_b=*/false);
}

// Load 8 consecutive fp32 from p, scale, convert to f16 into fr.h[base..base+7]
__device__ __forceinline__ void cvt8(Frag16& fr, int base, const float* p, float s) {
  const float4 f0 = *(const float4*)(p);
  const float4 f1 = *(const float4*)(p + 4);
  fr.h[base + 0] = (_Float16)(f0.x * s);
  fr.h[base + 1] = (_Float16)(f0.y * s);
  fr.h[base + 2] = (_Float16)(f0.z * s);
  fr.h[base + 3] = (_Float16)(f0.w * s);
  fr.h[base + 4] = (_Float16)(f1.x * s);
  fr.h[base + 5] = (_Float16)(f1.y * s);
  fr.h[base + 6] = (_Float16)(f1.z * s);
  fr.h[base + 7] = (_Float16)(f1.w * s);
}

// Async 16B copy global -> LDS (CDNA5, ASYNCcnt-tracked, GVS addressing)
__device__ __forceinline__ void async_b128(uint32_t ldsOff, uint32_t gOff,
                                           const float* base) {
  asm volatile("global_load_async_to_lds_b128 %0, %1, %2"
               :: "v"(ldsOff), "v"(gOff), "s"(base) : "memory");
}
__device__ __forceinline__ void wait_async0() {
  asm volatile("s_wait_asynccnt 0x0" ::: "memory");
}

// ---------------------------------------------------------------------------
// GEMM: Y[M,N] = X[M,K] @ W[K,N] + bias[N]   (fp32 in/out, f16 WMMA compute)
// 128-thread block = 4 waves computes a 64x64 output block; per 32-wide
// k-step the block async-copies a 64x32 A panel + 32x64 B panel into LDS.
// Wave w computes rows [w*16, w*16+16) x all 64 columns (4 WMMA tiles).
// ---------------------------------------------------------------------------
__global__ __launch_bounds__(128) void gemm_kernel(
    const float* __restrict__ X, const float* __restrict__ W,
    const float* __restrict__ bias, float* __restrict__ Y,
    int M, int N, int K) {
  __shared__ float Alds[BM * BK];   // [64][32] row-major, 8 KB
  __shared__ float Blds[BK * BN];   // [32][64] row-major, 8 KB

  const int t     = threadIdx.x;
  const int lane  = t & 31;
  const int l16   = lane & 15;
  const int half  = lane >> 4;       // lane half: 0 or 1
  const int koff8 = half * 8;        // C-layout row offset
  const int k16   = half * 16;       // B-layout K chunk offset
  const int w     = t >> 5;          // wave 0..3

  const int nblocks = N / BN;
  const int mbase = (blockIdx.x / nblocks) * BM;
  const int nbase = (blockIdx.x % nblocks) * BN;

  const uint32_t aLds = (uint32_t)(uintptr_t)Alds;
  const uint32_t bLds = (uint32_t)(uintptr_t)Blds;

  FragC c[4];
  #pragma unroll
  for (int n = 0; n < 4; ++n)
    #pragma unroll
    for (int i = 0; i < 8; ++i) c[n].f[i] = 0.0f;

  for (int k0 = 0; k0 < K; k0 += BK) {
    __syncthreads();  // previous panels fully consumed
    // Stage panels: 512 floats each per round x 4 rounds (float4 per thread)
    #pragma unroll
    for (int r = 0; r < 4; ++r) {
      const int flat = r * 128 + t;
      {  // A panel: row = flat/8, 8 float4 per 32-float row
        const int row = flat >> 3, c4 = (flat & 7) << 2;
        async_b128(aLds + (uint32_t)(row * BK + c4) * 4u,
                   (uint32_t)(((size_t)(mbase + row) * K + k0 + c4) * 4u), X);
      }
      {  // B panel: row = flat/16, 16 float4 per 64-float row
        const int row = flat >> 4, c4 = (flat & 15) << 2;
        async_b128(bLds + (uint32_t)(row * BN + c4) * 4u,
                   (uint32_t)(((size_t)(k0 + row) * N + nbase + c4) * 4u), W);
      }
    }
    wait_async0();
    __syncthreads();  // panels visible to all waves

    // A fragment for this wave (row = w*16 + l16)
    Frag16 a;
    const float* ar = Alds + (w * 16 + l16) * BK;
    cvt8(a, 0, ar + k16, 1.0f);
    cvt8(a, 8, ar + k16 + 8, 1.0f);

    #pragma unroll
    for (int n = 0; n < 4; ++n) {
      Frag16 bf;  // b[i] = Blds[k16 + i][n*16 + l16]
      const float* bp = Blds + k16 * BN + n * 16 + l16;
      #pragma unroll
      for (int i = 0; i < 16; ++i) bf.h[i] = (_Float16)bp[i * BN];
      c[n].v = wmma16(a.v, bf.v, c[n].v);
    }
  }

  #pragma unroll
  for (int n = 0; n < 4; ++n) {
    const int ncol = nbase + n * 16 + l16;
    const float bv = bias[ncol];
    const int rbase = mbase + w * 16 + koff8;
    #pragma unroll
    for (int r = 0; r < 8; ++r)
      Y[(size_t)(rbase + r) * N + ncol] = c[n].f[r] + bv;
  }
}

// ---------------------------------------------------------------------------
// Flash attention: one wave per (b, h, 16-row q-tile); streams key tiles of
// 32 with online softmax; O accumulated in f32 across HD=64 (4 C-tiles).
// qkv: [B, T, 3C] fp32 (q | k | v).  ybuf: [B, T, C] fp32.
// ---------------------------------------------------------------------------
__global__ __launch_bounds__(128) void attn_kernel(
    const float* __restrict__ qkv, float* __restrict__ ybuf) {
  const int lane  = threadIdx.x & 31;
  const int l16   = lane & 15;
  const int half  = lane >> 4;
  const int koff8 = half * 8;
  const int k16   = half * 16;
  const int wid   = blockIdx.x * 4 + (threadIdx.x >> 5);  // 0 .. 8191
  const int qt = wid & 127;
  const int h  = (wid >> 7) & (NHn - 1);
  const int b  = wid >> 11;
  const int qbase = qt << 4;

  __shared__ float pshare[4][16 * 32];        // P bounce buffer, per wave
  float* pbuf = pshare[threadIdx.x >> 5];

  // Q fragments (A layout), pre-scaled by 1/sqrt(HD) = 0.125
  Frag16 qA[2];
  {
    const float* qrow = qkv + (size_t)(b * Tn + qbase + l16) * C3n + h * HDn;
    #pragma unroll
    for (int cc = 0; cc < 2; ++cc) {
      const float* p = qrow + cc * 32;
      cvt8(qA[cc], 0, p + k16, 0.125f);
      cvt8(qA[cc], 8, p + k16 + 8, 0.125f);
    }
  }

  FragC o[4];
  #pragma unroll
  for (int n = 0; n < 4; ++n)
    #pragma unroll
    for (int i = 0; i < 8; ++i) o[n].f[i] = 0.0f;

  float mrow[8], lrow[8];
  #pragma unroll
  for (int r = 0; r < 8; ++r) { mrow[r] = -__builtin_inff(); lrow[r] = 0.0f; }

  const int nkt = (qbase + 16 + 31) >> 5;     // causal: keys <= qbase+15
  for (int kt = 0; kt < nkt; ++kt) {
    const int kb = kt << 5;

    // S(16x32) = Q @ K^T : two 16x16 C-tiles, each summed over HD in 2 steps
    FragC s[2];
    #pragma unroll
    for (int nh = 0; nh < 2; ++nh) {
      #pragma unroll
      for (int i = 0; i < 8; ++i) s[nh].f[i] = 0.0f;
      // B fragment of K^T: col = key (kb + nh*16 + l16); b[i] = d = cc*32+k16+i
      const float* kr = qkv + (size_t)(b * Tn + kb + nh * 16 + l16) * C3n
                        + Cn + h * HDn + k16;
      #pragma unroll
      for (int cc = 0; cc < 2; ++cc) {
        Frag16 kB;
        cvt8(kB, 0, kr + cc * 32, 1.0f);
        cvt8(kB, 8, kr + cc * 32 + 8, 1.0f);
        s[nh].v = wmma16(qA[cc].v, kB.v, s[nh].v);
      }
    }

    // Causal mask + online softmax (rows 0-7 in lanes 0-15; 8-15 in 16-31)
    const int n0 = kb + l16, n1 = n0 + 16;
    float fsc[8];
    #pragma unroll
    for (int r = 0; r < 8; ++r) {
      const int q = qbase + koff8 + r;
      if (n0 > q) s[0].f[r] = -__builtin_inff();
      if (n1 > q) s[1].f[r] = -__builtin_inff();
      float mx = fmaxf(s[0].f[r], s[1].f[r]);
      #pragma unroll
      for (int off = 8; off >= 1; off >>= 1)
        mx = fmaxf(mx, __shfl_xor(mx, off, 16));
      const float mn = fmaxf(mrow[r], mx);
      fsc[r] = __expf(mrow[r] - mn);
      mrow[r] = mn;
      const float p0 = __expf(s[0].f[r] - mn);
      const float p1 = __expf(s[1].f[r] - mn);
      s[0].f[r] = p0; s[1].f[r] = p1;
      float rs = p0 + p1;
      #pragma unroll
      for (int off = 8; off >= 1; off >>= 1)
        rs += __shfl_xor(rs, off, 16);
      lrow[r] = lrow[r] * fsc[r] + rs;
    }
    #pragma unroll
    for (int n = 0; n < 4; ++n)
      #pragma unroll
      for (int r = 0; r < 8; ++r) o[n].f[r] *= fsc[r];

    // P: C layout -> A layout via per-wave LDS bounce
    #pragma unroll
    for (int r = 0; r < 8; ++r) {
      const int row = koff8 + r;
      pbuf[row * 32 + l16]      = s[0].f[r];
      pbuf[row * 32 + 16 + l16] = s[1].f[r];
    }
    __asm__ volatile("s_wait_dscnt 0" ::: "memory");
    Frag16 pA;
    {
      const float* pr = pbuf + l16 * 32 + k16;
      cvt8(pA, 0, pr, 1.0f);
      cvt8(pA, 8, pr + 8, 1.0f);
    }

    // O += P @ V : 4 HD chunks of 16; V B fragment b[i] = key k16+i, col = d
    #pragma unroll
    for (int n = 0; n < 4; ++n) {
      Frag16 vB;
      const float* vp = qkv + (size_t)(b * Tn + kb + k16) * C3n
                        + 2 * Cn + h * HDn + n * 16 + l16;
      #pragma unroll
      for (int i = 0; i < 16; ++i) vB.h[i] = (_Float16)vp[(size_t)i * C3n];
      o[n].v = wmma16(pA.v, vB.v, o[n].v);
    }
  }

  float inv[8];
  #pragma unroll
  for (int r = 0; r < 8; ++r) inv[r] = 1.0f / lrow[r];
  #pragma unroll
  for (int n = 0; n < 4; ++n)
    #pragma unroll
    for (int r = 0; r < 8; ++r) {
      const size_t row = (size_t)(b * Tn + qbase + koff8 + r);
      ybuf[row * Cn + h * HDn + n * 16 + l16] = o[n].f[r] * inv[r];
    }
}

// ---------------------------------------------------------------------------
extern "C" void kernel_launch(void* const* d_in, const int* in_sizes, int n_in,
                              void* d_out, int out_size, void* d_ws, size_t ws_size,
                              hipStream_t stream) {
  const float* x      = (const float*)d_in[0];  // [B,T,C]
  const float* W_attn = (const float*)d_in[1];  // [C,3C]
  const float* b_attn = (const float*)d_in[2];  // [3C]
  const float* W_proj = (const float*)d_in[3];  // [C,C]
  const float* b_proj = (const float*)d_in[4];  // [C]
  float* out = (float*)d_out;                   // [B,T,C]

  float* qkv  = (float*)d_ws;                   // [B,T,3C]  96 MB
  float* ybuf = qkv + (size_t)BT * C3n;         // [B,T,C]   32 MB

  // 1) qkv = x @ W_attn + b_attn   (M=8192, N=3072, K=1024)
  {
    const int blocks = (BT / BM) * (C3n / BN);  // 128 * 48 = 6144
    gemm_kernel<<<blocks, 128, 0, stream>>>(x, W_attn, b_attn, qkv,
                                            BT, C3n, Cn);
  }
  // 2) flash attention -> ybuf
  {
    const int waves = Bn * NHn * (Tn / 16);     // 8192
    attn_kernel<<<waves / 4, 128, 0, stream>>>(qkv, ybuf);
  }
  // 3) out = ybuf @ W_proj + b_proj  (M=8192, N=1024, K=1024)
  {
    const int blocks = (BT / BM) * (Cn / BN);   // 128 * 16 = 2048
    gemm_kernel<<<blocks, 128, 0, stream>>>(ybuf, W_proj, b_proj, out,
                                            BT, Cn, Cn);
  }
}
